// TransformerLM_47098611368154
// MI455X (gfx1250) — matrix operand order repaired and checked by reading, not compiled, not run on hardware
//
#include <hip/hip_runtime.h>

// ---------------- model dims ----------------
#define SL   1024
#define SB   4
#define SD   1024
#define SH   16
#define SDH  64
#define SDFF 4096
#define SNL  6
#define NROW (SL*SB)
#define SCALE_ATTN 0.125f   // 1/sqrt(64)

typedef __attribute__((ext_vector_type(16))) __bf16 v16bf;
typedef __attribute__((ext_vector_type(8)))  float  v8f;
union Frag16 { uint4 q[2]; v16bf v; };

// ----- CDNA5 async global->LDS path (guarded: falls back to sync staging) -----
#if defined(__gfx1250__) && __has_builtin(__builtin_amdgcn_global_load_async_to_lds_b128) && \
    __has_builtin(__builtin_amdgcn_s_wait_asynccnt)
#define USE_ASYNC_LDS 1
#else
#define USE_ASYNC_LDS 0
#endif

#if USE_ASYNC_LDS
typedef int v4i_ __attribute__((vector_size(16)));
typedef __attribute__((address_space(1))) v4i_ glb_v4i;   // global <4 x i32>
typedef __attribute__((address_space(3))) v4i_ lds_v4i;   // LDS    <4 x i32>
static __device__ __forceinline__ glb_v4i* to_glb(const void* p) {
  return (glb_v4i*)(unsigned long long)p;
}
static __device__ __forceinline__ lds_v4i* to_lds(const void* p) {
  // generic LDS address keeps the byte offset in the low 32 bits on CDNA5
  return (lds_v4i*)(unsigned)(unsigned long long)p;
}
// copy 64B (one thread's slab slice): 4 x async B128, offsets apply to both sides
static __device__ __forceinline__ void async_cp64(const __bf16* g, __bf16* l) {
  glb_v4i* gp = to_glb(g); lds_v4i* lp = to_lds(l);
  __builtin_amdgcn_global_load_async_to_lds_b128(gp, lp, 0,  0);
  __builtin_amdgcn_global_load_async_to_lds_b128(gp, lp, 16, 0);
  __builtin_amdgcn_global_load_async_to_lds_b128(gp, lp, 32, 0);
  __builtin_amdgcn_global_load_async_to_lds_b128(gp, lp, 48, 0);
}
// copy 128B (one 64-elem bf16 row)
static __device__ __forceinline__ void async_cp128(const __bf16* g, __bf16* l) {
  glb_v4i* gp = to_glb(g); lds_v4i* lp = to_lds(l);
  __builtin_amdgcn_global_load_async_to_lds_b128(gp, lp, 0,   0);
  __builtin_amdgcn_global_load_async_to_lds_b128(gp, lp, 16,  0);
  __builtin_amdgcn_global_load_async_to_lds_b128(gp, lp, 32,  0);
  __builtin_amdgcn_global_load_async_to_lds_b128(gp, lp, 48,  0);
  __builtin_amdgcn_global_load_async_to_lds_b128(gp, lp, 64,  0);
  __builtin_amdgcn_global_load_async_to_lds_b128(gp, lp, 80,  0);
  __builtin_amdgcn_global_load_async_to_lds_b128(gp, lp, 96,  0);
  __builtin_amdgcn_global_load_async_to_lds_b128(gp, lp, 112, 0);
}
#endif

static __device__ __forceinline__ __bf16 f2bf(float f) {
  union { float f; unsigned u; } v; v.f = f;
  unsigned r = v.u + 0x7fffu + ((v.u >> 16) & 1u);
  union { unsigned short s; __bf16 b; } o; o.s = (unsigned short)(r >> 16);
  return o.b;
}
static __device__ __forceinline__ v8f wmma_bf(const v16bf a, const v16bf b, v8f c) {
  return __builtin_amdgcn_wmma_f32_16x16x32_bf16(false, a, false, b, (short)0, c, false, false);
}

// ---------------- elementwise: copy + cast ----------------
__global__ void k_cast_copy(const float* __restrict__ s, float* __restrict__ df,
                            __bf16* __restrict__ db, int n) {
  int i = blockIdx.x * 256 + threadIdx.x;
  if (i < n) { float v = s[i]; if (df) df[i] = v; db[i] = f2bf(v); }
}

// ---------------- weight transpose + cast: W[K,N] f32 -> Wt[N,K] bf16 ----------------
__global__ __launch_bounds__(256) void k_wtrans(const float* __restrict__ W,
                                                __bf16* __restrict__ Wt, int K, int N) {
  __shared__ float tile[32][33];
  const int kb = blockIdx.x * 32, nb = blockIdx.y * 32;
  const int tx = threadIdx.x & 31, ty = threadIdx.x >> 5;  // ty 0..7
  for (int r = ty; r < 32; r += 8) tile[r][tx] = W[(size_t)(kb + r) * N + nb + tx];
  __syncthreads();
  for (int r = ty; r < 32; r += 8) Wt[(size_t)(nb + r) * K + kb + tx] = f2bf(tile[tx][r]);
}

// ---------------- sinusoid positional embedding (bf16) ----------------
__global__ void k_pos(__bf16* __restrict__ peb) {
  int idx = blockIdx.x * 256 + threadIdx.x;
  if (idx >= SL * (SD / 2)) return;
  int i = idx / (SD / 2), f = idx % (SD / 2);
  float p = (float)(SL - 1 - i);
  float invf = __expf(-((float)(2 * f) / (float)SD) * __logf(10000.0f));
  float a = p * invf;
  peb[(size_t)i * SD + f]          = f2bf(__sinf(a));
  peb[(size_t)i * SD + SD / 2 + f] = f2bf(__cosf(a));
}

// ---------------- WMMA GEMM: C[M,N] = A[M,K](bf16) * Bt[N,K]^T(bf16) ----------------
#define GBM 128
#define GBN 128
#define GBK 64
#define GSTR 72
__global__ __launch_bounds__(256) void k_gemm(
    const __bf16* __restrict__ A, const __bf16* __restrict__ Bt,
    const float* __restrict__ bias, float* __restrict__ Cf,
    __bf16* __restrict__ Cb, int M, int N, int K, int relu) {
  const int tid = threadIdx.x, lane = tid & 31, wave = tid >> 5;
  const int bm = blockIdx.y * GBM, bn = blockIdx.x * GBN;
  const int wm = (wave >> 2) * 64, wn = (wave & 3) * 32;
  const int ml = lane & 15;
  const int kbA0 = (lane & 16) ? 16 : 0;   // A: contiguous-16 fragment layout
  const int kbB0 = (lane & 16) ? 16 : 0;   // B: contiguous-16 fragment layout
  const int lr = tid >> 1, lc = (tid & 1) * 4;
  const v8f z = {0.f,0.f,0.f,0.f,0.f,0.f,0.f,0.f};
  v8f acc[4][2];
  #pragma unroll
  for (int mt = 0; mt < 4; mt++)
    #pragma unroll
    for (int nt = 0; nt < 2; nt++) acc[mt][nt] = z;

  auto compute_tile = [&](const __bf16* Ab, const __bf16* Bb) {
    #pragma unroll
    for (int kc = 0; kc < 2; kc++) {
      const int kA = kc * 32 + kbA0, kB = kc * 32 + kbB0;
      Frag16 af[4], bf[2];
      #pragma unroll
      for (int mt = 0; mt < 4; mt++) {
        const __bf16* p = Ab + (wm + mt * 16 + ml) * GSTR + kA;
        af[mt].q[0] = *(const uint4*)p; af[mt].q[1] = *(const uint4*)(p + 8);
      }
      #pragma unroll
      for (int nt = 0; nt < 2; nt++) {
        const __bf16* p = Bb + (wn + nt * 16 + ml) * GSTR + kB;
        bf[nt].q[0] = *(const uint4*)p; bf[nt].q[1] = *(const uint4*)(p + 8);
      }
      #pragma unroll
      for (int mt = 0; mt < 4; mt++)
        #pragma unroll
        for (int nt = 0; nt < 2; nt++)
          acc[mt][nt] = wmma_bf(af[mt].v, bf[nt].v, acc[mt][nt]);
    }
  };

#if USE_ASYNC_LDS
  __shared__ __bf16 As[2][GBM * GSTR];
  __shared__ __bf16 Bs[2][GBN * GSTR];
  const __bf16* gA = A  + (size_t)(bm + lr) * K + lc * 8;
  const __bf16* gB = Bt + (size_t)(bn + lr) * K + lc * 8;
  __bf16* dA[2] = { As[0] + lr * GSTR + lc * 8, As[1] + lr * GSTR + lc * 8 };
  __bf16* dB[2] = { Bs[0] + lr * GSTR + lc * 8, Bs[1] + lr * GSTR + lc * 8 };
  const int ntiles = K / GBK;
  async_cp64(gA, dA[0]);
  async_cp64(gB, dB[0]);
  int t = 0;
  for (; t < ntiles - 1; t++) {
    async_cp64(gA + (size_t)(t + 1) * GBK, dA[(t + 1) & 1]);
    async_cp64(gB + (size_t)(t + 1) * GBK, dB[(t + 1) & 1]);
    __builtin_amdgcn_s_wait_asynccnt(8);   // tile t resident (in-order completion)
    __syncthreads();
    compute_tile(As[t & 1], Bs[t & 1]);
    __syncthreads();                       // all waves done before buffer reuse
  }
  __builtin_amdgcn_s_wait_asynccnt(0);
  __syncthreads();
  compute_tile(As[t & 1], Bs[t & 1]);
#else
  __shared__ __bf16 As1[GBM * GSTR];
  __shared__ __bf16 Bs1[GBN * GSTR];
  const uint4* gA = (const uint4*)(A  + (size_t)(bm + lr) * K) + lc;
  const uint4* gB = (const uint4*)(Bt + (size_t)(bn + lr) * K) + lc;
  uint4* sA = (uint4*)(As1 + lr * GSTR) + lc;
  uint4* sB = (uint4*)(Bs1 + lr * GSTR) + lc;
  for (int kk = 0; kk < K; kk += GBK) {
    uint4 a0 = gA[0], a1 = gA[1], a2 = gA[2], a3 = gA[3];
    uint4 b0 = gB[0], b1 = gB[1], b2 = gB[2], b3 = gB[3];
    gA += GBK / 8; gB += GBK / 8;
    __builtin_prefetch(gA, 0, 1);
    __builtin_prefetch(gB, 0, 1);
    __syncthreads();
    sA[0] = a0; sA[1] = a1; sA[2] = a2; sA[3] = a3;
    sB[0] = b0; sB[1] = b1; sB[2] = b2; sB[3] = b3;
    __syncthreads();
    compute_tile(As1, Bs1);
  }
#endif

  const int mh = (lane < 16) ? 0 : 8;
  #pragma unroll
  for (int nt = 0; nt < 2; nt++) {
    const int col = bn + wn + nt * 16 + ml;
    const float bv = bias ? bias[col] : 0.f;
    #pragma unroll
    for (int mt = 0; mt < 4; mt++) {
      const int r0 = bm + wm + mt * 16 + mh;
      #pragma unroll
      for (int v = 0; v < 8; v++) {
        float val = acc[mt][nt][v] + bv;
        if (relu) val = fmaxf(val, 0.f);
        size_t off = (size_t)(r0 + v) * N + col;
        if (Cf) Cf[off] = val;
        if (Cb) Cb[off] = f2bf(val);
      }
    }
  }
}

// ---------------- pack q(+u/+v bias), k, v^T, r into bf16 head-major layouts --------
__global__ void k_pack_qkvr(const float* __restrict__ q, const float* __restrict__ kv,
                            const float* __restrict__ r, const float* __restrict__ ub,
                            const float* __restrict__ vb,
                            __bf16* __restrict__ qu, __bf16* __restrict__ qv,
                            __bf16* __restrict__ kp, __bf16* __restrict__ vt,
                            __bf16* __restrict__ rp) {
  size_t idx = (size_t)blockIdx.x * 256 + threadIdx.x;
  if (idx >= (size_t)SB * SH * SL * SDH) return;
  int d = (int)(idx & (SDH - 1));
  size_t t = idx >> 6;
  int l = (int)(t & (SL - 1)); t >>= 10;
  int hh = (int)(t & (SH - 1)); int b = (int)(t >> 4);
  int row = l * SB + b, col = hh * SDH + d;
  size_t po = (((size_t)(b * SH + hh) * SL) + l) * SDH + d;
  float qq = q[(size_t)row * SD + col];
  qu[po] = f2bf(qq + ub[col]);
  qv[po] = f2bf(qq + vb[col]);
  kp[po] = f2bf(kv[(size_t)row * 2 * SD + col]);
  vt[(((size_t)(b * SH + hh) * SDH) + d) * SL + l] = f2bf(kv[(size_t)row * 2 * SD + SD + col]);
  if (b == 0) rp[((size_t)hh * SL + l) * SDH + d] = f2bf(r[(size_t)l * SD + col]);
}

// ---------------- fused flash attention with Transformer-XL rel-shift ----------------
#define ASTRK 72
__global__ __launch_bounds__(32) void k_attn(
    const __bf16* __restrict__ qu, const __bf16* __restrict__ qv,
    const __bf16* __restrict__ kp, const __bf16* __restrict__ vt,
    const __bf16* __restrict__ rp, __bf16* __restrict__ avb) {
  const int i0 = blockIdx.x * 16, hh = blockIdx.y, b = blockIdx.z;
  const int lane = threadIdx.x, ml = lane & 15;
  const int kbA0 = (lane & 16) ? 16 : 0;
  const int kbB0 = (lane & 16) ? 16 : 0;
  const int mh = (lane < 16) ? 0 : 8;
  __shared__ __bf16 Ks[32 * ASTRK];
  __shared__ __bf16 Rw[48 * ASTRK];
  __shared__ float Sac[16 * 34];
  __shared__ float Sbd[16 * 50];
  __shared__ __bf16 Pb[16 * 40];
  __shared__ float rsc[16];

  const v8f z = {0.f,0.f,0.f,0.f,0.f,0.f,0.f,0.f};
  v8f oacc[4];
  #pragma unroll
  for (int nt = 0; nt < 4; nt++) oacc[nt] = z;
  float m_i = -3.0e38f, l_i = 0.f;

  const size_t bh = (size_t)(b * SH + hh);
  const size_t qoff = (bh * SL + i0) * SDH;
  Frag16 aqu[2], aqv[2];
  #pragma unroll
  for (int kc = 0; kc < 2; kc++) {
    const __bf16* p = qu + qoff + (size_t)ml * SDH + kc * 32 + kbA0;
    aqu[kc].q[0] = *(const uint4*)p;  aqu[kc].q[1] = *(const uint4*)(p + 8);
    const __bf16* p2 = qv + qoff + (size_t)ml * SDH + kc * 32 + kbA0;
    aqv[kc].q[0] = *(const uint4*)p2; aqv[kc].q[1] = *(const uint4*)(p2 + 8);
  }
  const size_t kbase = bh * SL * SDH;
  const size_t vbase = bh * SDH * SL;
  const int jend = i0 + 15;
  for (int j0 = 0; j0 <= jend; j0 += 32) {
    const int base15 = SL - 16 - i0 + j0;   // r-window start
#if USE_ASYNC_LDS
    {
      int jr = j0 + lane; jr = jr > SL - 1 ? SL - 1 : jr;
      async_cp128(kp + kbase + (size_t)jr * SDH, Ks + lane * ASTRK);
    }
    for (int w = lane; w < 48; w += 32) {
      int ri = base15 + w; ri = ri < 0 ? 0 : (ri > SL - 1 ? SL - 1 : ri);
      async_cp128(rp + ((size_t)hh * SL + ri) * SDH, Rw + w * ASTRK);
    }
    __builtin_amdgcn_s_wait_asynccnt(0);
    __syncthreads();
#else
    {
      int jr = j0 + lane; jr = jr > SL - 1 ? SL - 1 : jr;
      const uint4* g = (const uint4*)(kp + kbase + (size_t)jr * SDH);
      uint4* s = (uint4*)(Ks + lane * ASTRK);
      #pragma unroll
      for (int t = 0; t < 8; t++) s[t] = g[t];
    }
    for (int w = lane; w < 48; w += 32) {
      int ri = base15 + w; ri = ri < 0 ? 0 : (ri > SL - 1 ? SL - 1 : ri);
      const uint4* g = (const uint4*)(rp + ((size_t)hh * SL + ri) * SDH);
      uint4* s = (uint4*)(Rw + w * ASTRK);
      #pragma unroll
      for (int t = 0; t < 8; t++) s[t] = g[t];
    }
    __syncthreads();
#endif
    v8f ac[2] = {z, z};
    v8f pd[3] = {z, z, z};
    #pragma unroll
    for (int kc = 0; kc < 2; kc++) {
      const int kB = kc * 32 + kbB0;
      Frag16 fb;
      #pragma unroll
      for (int nt = 0; nt < 2; nt++) {
        const __bf16* p = Ks + (nt * 16 + ml) * ASTRK + kB;
        fb.q[0] = *(const uint4*)p; fb.q[1] = *(const uint4*)(p + 8);
        ac[nt] = wmma_bf(aqu[kc].v, fb.v, ac[nt]);
      }
      #pragma unroll
      for (int nt = 0; nt < 3; nt++) {
        const __bf16* p = Rw + (nt * 16 + ml) * ASTRK + kB;
        fb.q[0] = *(const uint4*)p; fb.q[1] = *(const uint4*)(p + 8);
        pd[nt] = wmma_bf(aqv[kc].v, fb.v, pd[nt]);
      }
    }
    #pragma unroll
    for (int nt = 0; nt < 2; nt++)
      #pragma unroll
      for (int v = 0; v < 8; v++) Sac[(mh + v) * 34 + nt * 16 + ml] = ac[nt][v];
    #pragma unroll
    for (int nt = 0; nt < 3; nt++)
      #pragma unroll
      for (int v = 0; v < 8; v++) Sbd[(mh + v) * 50 + nt * 16 + ml] = pd[nt][v];
    __syncthreads();
    if (lane < 16) {               // online softmax, one row per lane
      const int i = i0 + lane;
      float mx = m_i;
      for (int jj = 0; jj < 32; jj++) {
        const int j = j0 + jj;
        if (j <= i) {
          float s = (Sac[lane * 34 + jj] + Sbd[lane * 50 + jj - lane + 15]) * SCALE_ATTN;
          mx = fmaxf(mx, s);
        }
      }
      const float corr = __expf(m_i - mx);
      float ls = l_i * corr;
      for (int jj = 0; jj < 32; jj++) {
        const int j = j0 + jj;
        float pv = 0.f;
        if (j <= i) {
          float s = (Sac[lane * 34 + jj] + Sbd[lane * 50 + jj - lane + 15]) * SCALE_ATTN;
          pv = __expf(s - mx);
        }
        ls += pv;
        Pb[lane * 40 + jj] = f2bf(pv);
      }
      m_i = mx; l_i = ls; rsc[lane] = corr;
    }
    __syncthreads();
    Frag16 ap;
    {
      const __bf16* p = Pb + ml * 40 + kbA0;
      ap.q[0] = *(const uint4*)p; ap.q[1] = *(const uint4*)(p + 8);
    }
    #pragma unroll
    for (int nt = 0; nt < 4; nt++) {
      #pragma unroll
      for (int v = 0; v < 8; v++) oacc[nt][v] *= rsc[mh + v];
      const __bf16* p = vt + vbase + (size_t)(nt * 16 + ml) * SL + j0 + kbB0;
      Frag16 fb; fb.q[0] = *(const uint4*)p; fb.q[1] = *(const uint4*)(p + 8);
      oacc[nt] = wmma_bf(ap.v, fb.v, oacc[nt]);
    }
    __syncthreads();
  }
  if (lane < 16) rsc[lane] = 1.f / l_i;
  __syncthreads();
  #pragma unroll
  for (int nt = 0; nt < 4; nt++)
    #pragma unroll
    for (int v = 0; v < 8; v++) {
      const int irow = i0 + mh + v;
      avb[(size_t)(irow * SB + b) * SD + hh * SDH + nt * 16 + ml] =
          f2bf(oacc[nt][v] * rsc[mh + v]);
    }
}

// ---------------- residual add + LayerNorm (writes f32 h and bf16 copy) -------------
__global__ __launch_bounds__(256) void k_add_ln(
    const float* __restrict__ h, const float* __restrict__ dlt,
    const float* __restrict__ g, const float* __restrict__ bb,
    float* __restrict__ ho, __bf16* __restrict__ hob) {
  __shared__ float red[256];
  const int row = blockIdx.x, tid = threadIdx.x;
  const float* hr = h + (size_t)row * SD;
  const float* dr = dlt + (size_t)row * SD;
  float loc[SD / 256];
  float s = 0.f;
  #pragma unroll
  for (int i = 0; i < SD / 256; i++) {
    float v = hr[tid + i * 256] + dr[tid + i * 256];
    loc[i] = v; s += v;
  }
  red[tid] = s; __syncthreads();
  for (int o = 128; o > 0; o >>= 1) { if (tid < o) red[tid] += red[tid + o]; __syncthreads(); }
  const float mean = red[0] * (1.f / SD);
  __syncthreads();
  float vs = 0.f;
  #pragma unroll
  for (int i = 0; i < SD / 256; i++) { float d = loc[i] - mean; vs += d * d; }
  red[tid] = vs; __syncthreads();
  for (int o = 128; o > 0; o >>= 1) { if (tid < o) red[tid] += red[tid + o]; __syncthreads(); }
  const float inv = rsqrtf(red[0] * (1.f / SD) + 1e-5f);
  #pragma unroll
  for (int i = 0; i < SD / 256; i++) {
    const int c = tid + i * 256;
    float o = (loc[i] - mean) * inv * g[c] + bb[c];
    ho[(size_t)row * SD + c] = o;
    hob[(size_t)row * SD + c] = f2bf(o);
  }
}

// ---------------- host driver ----------------
extern "C" void kernel_launch(void* const* d_in, const int* in_sizes, int n_in,
                              void* d_out, int out_size, void* d_ws, size_t ws_size,
                              hipStream_t stream) {
  (void)in_sizes; (void)n_in; (void)ws_size;
  const float* x    = (const float*)d_in[0];
  const float* Wq   = (const float*)d_in[1];
  const float* Wkv  = (const float*)d_in[2];
  const float* Wr   = (const float*)d_in[3];
  const float* Wo   = (const float*)d_in[4];
  const float* ln1g = (const float*)d_in[5];
  const float* ln1b = (const float*)d_in[6];
  const float* W1   = (const float*)d_in[7];
  const float* b1   = (const float*)d_in[8];
  const float* W2   = (const float*)d_in[9];
  const float* b2   = (const float*)d_in[10];
  const float* ln2g = (const float*)d_in[11];
  const float* ln2b = (const float*)d_in[12];
  const float* pbu  = (const float*)d_in[13];
  const float* pbv  = (const float*)d_in[14];

  char* wp = (char*)d_ws;
  auto alloc = [&](size_t bytes) -> void* {
    void* r = (void*)wp;
    wp += (bytes + 511) & ~(size_t)255;   // 256B align + slack for tile overreads
    return r;
  };
  __bf16* WqT  = (__bf16*)alloc((size_t)SNL * SD * SD * 2);
  __bf16* WkvT = (__bf16*)alloc((size_t)SNL * 2 * SD * SD * 2);
  __bf16* WrT  = (__bf16*)alloc((size_t)SNL * SD * SD * 2);
  __bf16* WoT  = (__bf16*)alloc((size_t)SNL * SD * SD * 2);
  __bf16* W1T  = (__bf16*)alloc((size_t)SNL * SDFF * SD * 2);
  __bf16* W2T  = (__bf16*)alloc((size_t)SNL * SD * SDFF * 2);
  float*  hbuf = (float*)alloc((size_t)NROW * SD * 4);
  __bf16* hb   = (__bf16*)alloc((size_t)NROW * SD * 2);
  float*  qb   = (float*)alloc((size_t)NROW * SD * 4);
  float*  kvb  = (float*)alloc((size_t)NROW * 2 * SD * 4);
  float*  rb   = (float*)alloc((size_t)SL * SD * 4);
  __bf16* peb  = (__bf16*)alloc((size_t)SL * SD * 2);
  const size_t qkvN = (size_t)SB * SH * SL * SDH;
  __bf16* qup  = (__bf16*)alloc(qkvN * 2);
  __bf16* qvp  = (__bf16*)alloc(qkvN * 2);
  __bf16* kpp  = (__bf16*)alloc(qkvN * 2);
  __bf16* vtp  = (__bf16*)alloc(qkvN * 2);
  __bf16* rpp  = (__bf16*)alloc((size_t)SH * SL * SDH * 2);
  __bf16* avb  = (__bf16*)alloc((size_t)NROW * SD * 2);
  float*  ob   = (float*)alloc((size_t)NROW * SD * 4);
  __bf16* f1b  = (__bf16*)alloc((size_t)NROW * SDFF * 2);

  auto wtrans = [&](const float* W, __bf16* Wt, int K, int N) {
    k_wtrans<<<dim3(K / 32, N / 32), 256, 0, stream>>>(W, Wt, K, N);
  };
  for (int l = 0; l < SNL; l++) {
    wtrans(Wq  + (size_t)l * SD * SD,     WqT  + (size_t)l * SD * SD,     SD, SD);
    wtrans(Wkv + (size_t)l * SD * 2 * SD, WkvT + (size_t)l * 2 * SD * SD, SD, 2 * SD);
    wtrans(Wr  + (size_t)l * SD * SD,     WrT  + (size_t)l * SD * SD,     SD, SD);
    wtrans(Wo  + (size_t)l * SD * SD,     WoT  + (size_t)l * SD * SD,     SD, SD);
    wtrans(W1  + (size_t)l * SD * SDFF,   W1T  + (size_t)l * SDFF * SD,   SD, SDFF);
    wtrans(W2  + (size_t)l * SDFF * SD,   W2T  + (size_t)l * SD * SDFF,   SDFF, SD);
  }
  k_cast_copy<<<(NROW * SD) / 256, 256, 0, stream>>>(x, hbuf, hb, NROW * SD);
  k_pos<<<(SL * SD / 2) / 256, 256, 0, stream>>>(peb);

  auto gemm = [&](const __bf16* A, const __bf16* Bt, const float* bias,
                  float* Cf, __bf16* Cb, int M, int N, int K, int relu) {
    k_gemm<<<dim3(N / 128, M / 128), 256, 0, stream>>>(A, Bt, bias, Cf, Cb, M, N, K, relu);
  };

  for (int l = 0; l < SNL; l++) {
    gemm(hb,  WqT  + (size_t)l * SD * SD,     nullptr, qb,  nullptr, NROW, SD,     SD,   0);
    gemm(hb,  WkvT + (size_t)l * 2 * SD * SD, nullptr, kvb, nullptr, NROW, 2 * SD, SD,   0);
    gemm(peb, WrT  + (size_t)l * SD * SD,     nullptr, rb,  nullptr, SL,   SD,     SD,   0);
    k_pack_qkvr<<<(int)(qkvN / 256), 256, 0, stream>>>(qb, kvb, rb, pbu, pbv,
                                                       qup, qvp, kpp, vtp, rpp);
    k_attn<<<dim3(SL / 16, SH, SB), 32, 0, stream>>>(qup, qvp, kpp, vtp, rpp, avb);
    gemm(avb, WoT + (size_t)l * SD * SD, nullptr, ob, nullptr, NROW, SD, SD, 0);
    k_add_ln<<<NROW, 256, 0, stream>>>(hbuf, ob, ln1g + (size_t)l * SD,
                                       ln1b + (size_t)l * SD, hbuf, hb);
    gemm(hb,  W1T + (size_t)l * SDFF * SD, b1 + (size_t)l * SDFF, nullptr, f1b,
         NROW, SDFF, SD, 1);
    gemm(f1b, W2T + (size_t)l * SD * SDFF, b2 + (size_t)l * SD, ob, nullptr,
         NROW, SD, SDFF, 0);
    k_add_ln<<<NROW, 256, 0, stream>>>(hbuf, ob, ln2g + (size_t)l * SD,
                                       ln2b + (size_t)l * SD, hbuf, hb);
  }
  (void)hipMemcpyAsync(d_out, hbuf, (size_t)out_size * sizeof(float),
                       hipMemcpyDeviceToDevice, stream);
}
